// RadianceField_43731357007945
// MI455X (gfx1250) — matrix operand order repaired and checked by reading, not compiled, not run on hardware
//
#include <hip/hip_runtime.h>
#include <hip/hip_bf16.h>

typedef __attribute__((ext_vector_type(16))) _Float16 v16h;
typedef __attribute__((ext_vector_type(8)))  float    v8f;

#define NPTS      64      // points per block
#define NTHREADS  128     // 4 waves * 32 lanes
#define HASH_T    524288  // 2^19
#define HASH_L    16

// resolutions: floor(16 * exp(l * ln(128)/15))
__device__ __constant__ float c_res[HASH_L] = {
    16.f, 22.f, 30.f, 42.f, 58.f, 80.f, 111.f, 153.f,
    212.f, 294.f, 406.f, 561.f, 776.f, 1072.f, 1482.f, 2048.f
};

// ---------------- WMMA fragment helpers (wave32, 16-bit A/B layouts per CDNA5 ISA 7.12.2) ----

__device__ inline v8f wmma_f16(v16h a, v16h b, v8f c) {
    // (neg_a, A, neg_b, B, c_mod, C, reuse_a, reuse_b)
    return __builtin_amdgcn_wmma_f32_16x16x32_f16(false, a, false, b, (short)0, c, false, false);
}

// A: 16x32 f16. lane L: M = L%15? no: M = L&15, half = L>>4.
// VGPR v<4: K = half*8 + 2v, {+0,+1}; v>=4: K = 16 + half*8 + 2*(v-4), {+0,+1}.
__device__ inline v16h load_a(const _Float16* __restrict__ base, int stride, int lane, int kbase) {
    int m = lane & 15, h = lane >> 4;
    const _Float16* row = base + m * stride + kbase;
    union { v16h v; unsigned u[8]; } f;
#pragma unroll
    for (int i = 0; i < 8; ++i) {
        int k = (i < 4) ? (h * 8 + 2 * i) : (16 + h * 8 + 2 * (i - 4));
        f.u[i] = *(const unsigned*)(row + k);
    }
    return f.v;
}

// B: 32x16 f16, weights stored transposed in LDS as [N][Kpad].
// lane L: N = L&15, khalf = L>>4; VGPR v<4: K = khalf*8 + 2v, v>=4: K = 16 + khalf*8 + 2*(v-4), {+0,+1}.
__device__ inline v16h load_b(const _Float16* __restrict__ wt, int kpad, int lane, int kbase) {
    int n = lane & 15, kh = lane >> 4;
    const _Float16* row = wt + n * kpad + kbase;
    union { v16h v; unsigned u[8]; } f;
#pragma unroll
    for (int i = 0; i < 8; ++i) f.u[i] = *(const unsigned*)(row + ((i < 4) ? (kh * 8 + 2 * i) : (16 + kh * 8 + 2 * (i - 4))));
    return f.v;
}

// C/D: lane L, VGPR r -> (M = r + 8*(L>>4), N = L&15)
__device__ inline void store_d_relu(v8f c, _Float16* __restrict__ base, int stride, int lane) {
    int n = lane & 15, hi = lane >> 4;
#pragma unroll
    for (int r = 0; r < 8; ++r) {
        float v = fmaxf(c[r], 0.f);
        base[(r + 8 * hi) * stride + n] = (_Float16)v;
    }
}

__device__ inline v8f bias_c(const float* __restrict__ bias, int lane) {
    float b = bias[lane & 15];
    v8f c;
#pragma unroll
    for (int r = 0; r < 8; ++r) c[r] = b;
    return c;
}

// ---------------------------------------------------------------------------------------------

__global__ __launch_bounds__(NTHREADS)
void nerf_fused_kernel(const float* __restrict__ pos,    // [N,3]
                       const float* __restrict__ dirs,   // [B,3]
                       const float* __restrict__ table,  // [L,T,2]
                       const float* __restrict__ w1,  const float* __restrict__ b1,
                       const float* __restrict__ w2,  const float* __restrict__ b2,
                       const float* __restrict__ w0r, const float* __restrict__ b0r,
                       const float* __restrict__ w1r, const float* __restrict__ b1r,
                       const float* __restrict__ w2r, const float* __restrict__ b2r,
                       float* __restrict__ out, int npoints) {
    // ----- LDS -----
    __shared__ _Float16 sW1t [64 * 32];    // base_w1^T  [N=64][K=32]
    __shared__ _Float16 sW2t [16 * 64];    // base_w2^T  [N=16][K=64]   (N 15->16 pad)
    __shared__ _Float16 sW0t [64 * 64];    // rgb_w0^T   [N=64][K=64]   (K 42->64 pad)
    __shared__ _Float16 sW1rt[64 * 128];   // rgb_w1^T   [N=64][K=128]  (K 106->128 pad)
    __shared__ _Float16 sW2rt[16 * 64];    // rgb_w2^T   [N=16][K=64]   (N 3->16 pad)
    __shared__ float sB1[64], sB2[16], sB0[64], sB1r[64], sB2r[16];
    // activation region: enc = sAct[0..2047] (64x32), h1 = sAct[2048..6143] (64x64)
    //                    h2 aliases sAct[0..4095] (64x64) after h1 dies
    __shared__ _Float16 sAct[6144];
    // rinBig: [64 pts][128]: cols 0..63 = h (rgb layer0 out), 64..78 geo, 79..105 direnc, 106..127 zero
    __shared__ _Float16 sRin[64 * 128];

    const int tid  = threadIdx.x;
    const int lane = tid & 31;
    const int wave = tid >> 5;

    // ----- load + transpose + pad weights into LDS (f32 -> f16) -----
    for (int i = tid; i < 64 * 32; i += NTHREADS) {
        int n = i >> 5, k = i & 31;
        sW1t[i] = (_Float16)w1[k * 64 + n];
    }
    for (int i = tid; i < 16 * 64; i += NTHREADS) {
        int n = i >> 6, k = i & 63;
        sW2t[i] = (_Float16)((n < 15) ? w2[k * 15 + n] : 0.f);
    }
    for (int i = tid; i < 64 * 64; i += NTHREADS) {
        int n = i >> 6, k = i & 63;
        sW0t[i] = (_Float16)((k < 42) ? w0r[k * 64 + n] : 0.f);
    }
    for (int i = tid; i < 64 * 128; i += NTHREADS) {
        int n = i >> 7, k = i & 127;
        sW1rt[i] = (_Float16)((k < 106) ? w1r[k * 64 + n] : 0.f);
    }
    for (int i = tid; i < 16 * 64; i += NTHREADS) {
        int n = i >> 6, k = i & 63;
        sW2rt[i] = (_Float16)((n < 3) ? w2r[k * 3 + n] : 0.f);
    }
    if (tid < 64) { sB1[tid] = b1[tid]; sB0[tid] = b0r[tid]; sB1r[tid] = b1r[tid]; }
    if (tid < 16) { sB2[tid]  = (tid < 15) ? b2[tid]  : 0.f;
                    sB2r[tid] = (tid < 3)  ? b2r[tid] : 0.f; }

    // ----- Phase A: contraction + hash encode + direction encode (2 threads / point) -----
    {
        const int lp   = tid >> 1;            // local point 0..63
        const int part = tid & 1;
        int gp = blockIdx.x * NPTS + lp;
        int gpc = (gp < npoints) ? gp : (npoints - 1);

        float x = pos[(size_t)gpc * 3 + 0];
        float y = pos[(size_t)gpc * 3 + 1];
        float z = pos[(size_t)gpc * 3 + 2];
        float xn0 = x * (1.f / 50.f), xn1 = y * (1.f / 50.f), xn2 = z * (1.f / 10.f);
        float mag = fmaxf(fmaxf(fabsf(xn0), fabsf(xn1)), fmaxf(fabsf(xn2), 1e-6f));
        float s0, s1, s2;
        if (mag <= 1.f) {
            s0 = xn0 * 0.5f; s1 = xn1 * 0.5f; s2 = xn2 * 0.5f;
        } else {
            float fac = (1.f - 0.5f / mag) / mag;
            s0 = xn0 * fac; s1 = xn1 * fac; s2 = xn2 * fac;
        }
        float xc0 = (s0 + 1.f) * 0.5f, xc1 = (s1 + 1.f) * 0.5f, xc2 = (s2 + 1.f) * 0.5f;

        // 8 hash levels per thread
        for (int l = part * 8; l < part * 8 + 8; ++l) {
            float res = c_res[l];
            float xs0 = xc0 * res, xs1 = xc1 * res, xs2 = xc2 * res;
            float f0 = floorf(xs0), f1 = floorf(xs1), f2 = floorf(xs2);
            float wa = xs0 - f0, wb = xs1 - f1, wc = xs2 - f2;
            unsigned u0 = (unsigned)(int)f0, u1 = (unsigned)(int)f1, u2 = (unsigned)(int)f2;
            const float* tab = table + (size_t)l * (size_t)(HASH_T * 2);
            float acc0 = 0.f, acc1 = 0.f;
#pragma unroll
            for (int c = 0; c < 8; ++c) {
                unsigned cx = u0 + ((c >> 2) & 1);
                unsigned cy = u1 + ((c >> 1) & 1);
                unsigned cz = u2 + (c & 1);
                unsigned hsh = cx ^ (cy * 2654435761u) ^ (cz * 805459861u);
                unsigned idx = hsh & (unsigned)(HASH_T - 1);
                float cw = ((c & 4) ? wa : 1.f - wa) *
                           ((c & 2) ? wb : 1.f - wb) *
                           ((c & 1) ? wc : 1.f - wc);
                acc0 += cw * tab[(size_t)idx * 2 + 0];
                acc1 += cw * tab[(size_t)idx * 2 + 1];
            }
            sAct[lp * 32 + l * 2 + 0] = (_Float16)acc0;
            sAct[lp * 32 + l * 2 + 1] = (_Float16)acc1;
        }

        // direction encoding (per-ray, S=128) -> rinBig cols 79..105, zeros 106..127
        int ray = gpc >> 7;
        float d0 = dirs[(size_t)ray * 3 + 0];
        float d1 = dirs[(size_t)ray * 3 + 1];
        float d2 = dirs[(size_t)ray * 3 + 2];
        float de[27];
        de[0] = d0; de[1] = d1; de[2] = d2;
#pragma unroll
        for (int j = 0; j < 4; ++j) {
            float fr = (float)(1 << j);
            de[3 + j * 3 + 0]  = sinf(d0 * fr);
            de[3 + j * 3 + 1]  = sinf(d1 * fr);
            de[3 + j * 3 + 2]  = sinf(d2 * fr);
            de[15 + j * 3 + 0] = cosf(d0 * fr);
            de[15 + j * 3 + 1] = cosf(d1 * fr);
            de[15 + j * 3 + 2] = cosf(d2 * fr);
        }
        for (int j = part; j < 49; j += 2)
            sRin[lp * 128 + 79 + j] = (_Float16)((j < 27) ? de[j] : 0.f);
    }
    __syncthreads();

    // ----- WMMA MLP chain: wave handles 16 points (rows rowbase..rowbase+15) -----
    const int rowbase = wave * 16;
    const _Float16* encW = sAct + rowbase * 32;
    _Float16* h1W = sAct + 2048 + rowbase * 64;

    // Layer 1: enc[16x32] @ w1 -> relu -> h1[16x64]
#pragma unroll
    for (int nt = 0; nt < 4; ++nt) {
        v8f c = bias_c(sB1 + nt * 16, lane);
        v16h a = load_a(encW, 32, lane, 0);
        v16h b = load_b(sW1t + nt * 16 * 32, 32, lane, 0);
        c = wmma_f16(a, b, c);
        store_d_relu(c, h1W + nt * 16, 64, lane);
    }
    __syncthreads();

    // Layer 2: h1[16x64] @ w2 -> geo[16x15] (no relu); density from col 0
    {
        v8f c = bias_c(sB2, lane);
#pragma unroll
        for (int kc = 0; kc < 64; kc += 32) {
            v16h a = load_a(sAct + 2048 + rowbase * 64, 64, lane, kc);
            v16h b = load_b(sW2t, 64, lane, kc);
            c = wmma_f16(a, b, c);
        }
        int n = lane & 15, hi = lane >> 4;
        if (n == 0) {
#pragma unroll
            for (int r = 0; r < 8; ++r) {
                int gp2 = blockIdx.x * NPTS + rowbase + r + 8 * hi;
                if (gp2 < npoints) {
                    float g = fminf(fmaxf(c[r] - 1.f, -15.f), 15.f);
                    out[(size_t)gp2 * 4 + 0] = expf(g);
                }
            }
        }
        if (n < 15) {
#pragma unroll
            for (int r = 0; r < 8; ++r)
                sRin[(rowbase + r + 8 * hi) * 128 + 64 + n] = (_Float16)c[r];
        }
    }
    __syncthreads();

    // RGB layer 0: rin[16x42(->64)] @ rgb_w0 -> relu -> h in rinBig cols 0..63
#pragma unroll
    for (int nt = 0; nt < 4; ++nt) {
        v8f c = bias_c(sB0 + nt * 16, lane);
#pragma unroll
        for (int kc = 0; kc < 64; kc += 32) {
            v16h a = load_a(sRin + rowbase * 128 + 64, 128, lane, kc);
            v16h b = load_b(sW0t + nt * 16 * 64, 64, lane, kc);
            c = wmma_f16(a, b, c);
        }
        store_d_relu(c, sRin + rowbase * 128 + nt * 16, 128, lane);
    }
    __syncthreads();

    // RGB layer 1 (skip concat): [h|rin][16x106(->128)] @ rgb_w1 -> relu -> h2 (aliases sAct)
#pragma unroll
    for (int nt = 0; nt < 4; ++nt) {
        v8f c = bias_c(sB1r + nt * 16, lane);
#pragma unroll
        for (int kc = 0; kc < 128; kc += 32) {
            v16h a = load_a(sRin + rowbase * 128, 128, lane, kc);
            v16h b = load_b(sW1rt + nt * 16 * 128, 128, lane, kc);
            c = wmma_f16(a, b, c);
        }
        store_d_relu(c, sAct + rowbase * 64 + nt * 16, 64, lane);
    }
    __syncthreads();

    // RGB layer 2: h2[16x64] @ rgb_w2 -> sigmoid -> out[...,1..3]
    {
        v8f c = bias_c(sB2r, lane);
#pragma unroll
        for (int kc = 0; kc < 64; kc += 32) {
            v16h a = load_a(sAct + rowbase * 64, 64, lane, kc);
            v16h b = load_b(sW2rt, 64, lane, kc);
            c = wmma_f16(a, b, c);
        }
        int n = lane & 15, hi = lane >> 4;
        if (n < 3) {
#pragma unroll
            for (int r = 0; r < 8; ++r) {
                int gp2 = blockIdx.x * NPTS + rowbase + r + 8 * hi;
                if (gp2 < npoints)
                    out[(size_t)gp2 * 4 + 1 + n] = 1.f / (1.f + expf(-c[r]));
            }
        }
    }
}

extern "C" void kernel_launch(void* const* d_in, const int* in_sizes, int n_in,
                              void* d_out, int out_size, void* d_ws, size_t ws_size,
                              hipStream_t stream) {
    const float* pos   = (const float*)d_in[0];
    const float* dirs  = (const float*)d_in[1];
    const float* table = (const float*)d_in[2];
    const float* w1  = (const float*)d_in[3];
    const float* b1  = (const float*)d_in[4];
    const float* w2  = (const float*)d_in[5];
    const float* b2  = (const float*)d_in[6];
    const float* w0r = (const float*)d_in[7];
    const float* b0r = (const float*)d_in[8];
    const float* w1r = (const float*)d_in[9];
    const float* b1r = (const float*)d_in[10];
    const float* w2r = (const float*)d_in[11];
    const float* b2r = (const float*)d_in[12];
    float* out = (float*)d_out;

    int npoints = in_sizes[0] / 3;                 // 524288
    int blocks  = (npoints + NPTS - 1) / NPTS;     // 8192

    nerf_fused_kernel<<<blocks, NTHREADS, 0, stream>>>(
        pos, dirs, table, w1, b1, w2, b2, w0r, b0r, w1r, b1r, w2r, b2r,
        out, npoints);
}
